// SUPRAParallelGatedTrue_58102317580755
// MI455X (gfx1250) — matrix operand order repaired and checked by reading, not compiled, not run on hardware
//
#include <hip/hip_runtime.h>
#include <cstdint>
#include <cstddef>

// Problem constants (from reference)
#define B_ 4
#define T_ 4096
#define D_ 1024
static constexpr float EPS_ = 1e-6f;

// ---- POD vector types ----
typedef __attribute__((ext_vector_type(16))) __bf16    v16bf;
typedef __attribute__((ext_vector_type(8)))  float     v8f;
typedef __attribute__((ext_vector_type(4)))  unsigned  u32x4;
typedef __attribute__((ext_vector_type(4)))  int       i32x4;

union FragU { u32x4 q[2]; v16bf v; };

// ---- async global->LDS copy (CDNA5 ASYNCcnt path), with safe fallback ----
#if defined(__HIP_DEVICE_COMPILE__) && __has_builtin(__builtin_amdgcn_global_load_async_to_lds_b128)
#define ASYNC_LDS 1
#endif

__device__ __forceinline__ void cp16_to_lds(unsigned short* lds, const unsigned short* g) {
#ifdef ASYNC_LDS
    __builtin_amdgcn_global_load_async_to_lds_b128(
        (__attribute__((address_space(1))) i32x4*)(unsigned short*)g,
        (__attribute__((address_space(3))) i32x4*)lds, 0, 0);
#else
    *(u32x4*)lds = *(const u32x4*)g;
#endif
}

__device__ __forceinline__ void cp_wait_all() {
#ifdef ASYNC_LDS
#if __has_builtin(__builtin_amdgcn_s_wait_asynccnt)
    __builtin_amdgcn_s_wait_asynccnt(0);
#endif
#endif
}

// round-to-nearest-even f32 -> bf16 bits
__device__ __forceinline__ unsigned short f2bf(float f) {
    unsigned u = __float_as_uint(f);
    unsigned r = u + 0x7fffu + ((u >> 16) & 1u);
    return (unsigned short)(r >> 16);
}

__device__ __forceinline__ v16bf ld_frag(const unsigned short* p0, const unsigned short* p1) {
    FragU f;
    f.q[0] = *(const u32x4*)p0;
    f.q[1] = *(const u32x4*)p1;
    return f.v;
}

// A-matrix fragment (16x32 bf16, row-major source, row = lane%16):
//   VGPR0..3 <- K = kbase + h*8 + {0..7};  VGPR4..7 <- K = kbase + 16 + h*8 + {0..7}
__device__ __forceinline__ v16bf ld_afrag(const unsigned short* base, int ld, int kbase, int lane) {
    const int m = lane & 15, h = (lane >> 4) & 1;
    const unsigned short* p0 = base + (size_t)m * ld + kbase + h * 8;
    return ld_frag(p0, p0 + 16);
}

// B-matrix fragment (32x16 bf16) from source stored row-major by N (src[n][k]):
//   lane n = lane%16, half h: K = kbase + h*8 + {0..7}, then kbase + 16 + h*8 + {0..7}
__device__ __forceinline__ v16bf ld_bfrag(const unsigned short* base, size_t ld, int kbase, int lane) {
    const int n = lane & 15, h = (lane >> 4) & 1;
    const unsigned short* p = base + (size_t)n * ld + kbase + h * 8;
    return ld_frag(p, p + 16);
}

__device__ __forceinline__ v8f wmma_bf16(v16bf a, v16bf b, v8f c) {
    return __builtin_amdgcn_wmma_f32_16x16x32_bf16(false, a, false, b, (short)0, c, false, false);
}

// ---------------- Stage 0: f32 -> bf16 convert ----------------
__global__ void cvt_f32_bf16_kernel(const float* __restrict__ in, unsigned short* __restrict__ out, int n) {
    for (int i = blockIdx.x * blockDim.x + threadIdx.x; i < n; i += gridDim.x * blockDim.x)
        out[i] = f2bf(in[i]);
}

// ---------------- Stage 1: four fused projections ----------------
// y = act(x @ W^T + b). blockIdx.z selects projection.
// Each wave computes a 16x64 output tile (4 WMMAs per K-step, A-fragment reused).
__global__ __launch_bounds__(256) void proj_kernel(
    const unsigned short* __restrict__ xb,
    const unsigned short* __restrict__ Wqb, const unsigned short* __restrict__ Wkb,
    const unsigned short* __restrict__ Wvb, const unsigned short* __restrict__ Wab,
    const float* __restrict__ bq, const float* __restrict__ bk,
    const float* __restrict__ bv, const float* __restrict__ ba,
    unsigned short* __restrict__ q_out,   // bf16 [B*T, D]  (relu)
    float*          __restrict__ k_out,   // f32  [B*T, D]  (relu)
    unsigned short* __restrict__ vt_out,  // bf16 [B, D, T] (identity, transposed)
    float*          __restrict__ la_out)  // f32  [B*T, D]  log(sigmoid+1e-6)
{
    __shared__ __align__(16) unsigned short xs[16 * D_];
    const int tid = threadIdx.x;
    const int R0  = blockIdx.x * 16;          // global row tile over flattened B*T
    const int z   = blockIdx.z;

    // stage x tile (16 x 1024 bf16 = 32 KB) into LDS
    {
        const unsigned short* src = xb + (size_t)R0 * D_;
        for (int i = tid; i < 16 * D_ / 8; i += 256)
            cp16_to_lds(xs + (size_t)i * 8, src + (size_t)i * 8);
        cp_wait_all();
    }
    __syncthreads();

    const int wave = tid >> 5, lane = tid & 31;
    const int jt0 = (blockIdx.y * 8 + wave) * 64;   // this wave's 64 output columns

    const unsigned short* W = (z == 0) ? Wqb : (z == 1) ? Wkb : (z == 2) ? Wvb : Wab;
    const float* bias       = (z == 0) ? bq  : (z == 1) ? bk  : (z == 2) ? bv  : ba;

    const v8f vz = {0.f, 0.f, 0.f, 0.f, 0.f, 0.f, 0.f, 0.f};
    v8f acc[4];
    for (int c4 = 0; c4 < 4; ++c4) acc[c4] = vz;

    for (int k = 0; k < D_; k += 32) {
        v16bf a = ld_afrag(xs, D_, k, lane);
        #pragma unroll
        for (int c4 = 0; c4 < 4; ++c4) {
            const unsigned short* Wrow = W + (size_t)(jt0 + c4 * 16) * D_;
            v16bf b = ld_bfrag(Wrow, D_, k, lane);
            acc[c4] = wmma_bf16(a, b, acc[c4]);
        }
    }

    const int n = lane & 15, h = (lane >> 4) & 1;
    for (int c4 = 0; c4 < 4; ++c4) {
        const int j = jt0 + c4 * 16 + n;
        const float bj = bias[j];
        for (int r = 0; r < 8; ++r) {
            const int row = r + h * 8;
            const int gr  = R0 + row;             // flattened (b*T + t)
            const float y = acc[c4][r] + bj;
            if (z == 0) {
                q_out[(size_t)gr * D_ + j] = f2bf(fmaxf(y, 0.f));
            } else if (z == 1) {
                k_out[(size_t)gr * D_ + j] = fmaxf(y, 0.f);
            } else if (z == 2) {
                const int bb = gr >> 12;          // / T_
                const int t  = gr & (T_ - 1);
                vt_out[((size_t)bb * D_ + j) * T_ + t] = f2bf(y);
            } else {
                const float s = 1.f / (1.f + __expf(-y));
                la_out[(size_t)gr * D_ + j] = __logf(s + 1e-6f);
            }
        }
    }
}

// ---------------- Stage 2: reverse scan of log-gates, weighted k ----------------
// wk[b,t,d] = k[b,t,d] * exp(sum_{u>=t} log(A+1e-6))
__global__ __launch_bounds__(256) void scan_kernel(
    const float* __restrict__ kf, const float* __restrict__ la,
    unsigned short* __restrict__ wkb)
{
    const int d = blockIdx.x * 256 + threadIdx.x;
    const int b = blockIdx.y;
    const size_t base = (size_t)b * T_ * D_ + d;
    float acc = 0.f;
    for (int t = T_ - 1; t >= 0; --t) {
        const size_t o = base + (size_t)t * D_;
        acc += la[o];
        wkb[o] = f2bf(kf[o] * __expf(acc));
    }
}

// ---------------- Stage 3: causal attention ----------------
// out[t,:] = (sum_{s<=t} (q_t . wk_s) v_s) / (sum_{s<=t} q_t . wk_s + EPS)
// 256-key chunks; each wave computes two 16x16 score tiles (shared q A-fragment),
// then consumes the chunk against its own 128 output columns.
__global__ __launch_bounds__(256) void attn_kernel(
    const unsigned short* __restrict__ qb,   // bf16 [B*T, D]
    const unsigned short* __restrict__ wkb,  // bf16 [B*T, D]
    const unsigned short* __restrict__ vt,   // bf16 [B, D, T]
    float* __restrict__ out)                 // f32  [B, T, D]
{
    __shared__ __align__(16) unsigned short qs[16 * D_];   // query tile (32 KB)
    __shared__ __align__(16) unsigned short ss[16 * 256];  // bf16 scores, one key chunk (8 KB)
    __shared__ float denom_s[16];

    const int tid = threadIdx.x;
    const int b = blockIdx.y, tt = blockIdx.x;
    const int t0 = tt * 16, t_hi = t0 + 15;

    {
        const unsigned short* src = qb + ((size_t)b * T_ + t0) * D_;
        for (int i = tid; i < 16 * D_ / 8; i += 256)
            cp16_to_lds(qs + (size_t)i * 8, src + (size_t)i * 8);
        cp_wait_all();
    }
    if (tid < 16) denom_s[tid] = 0.f;
    __syncthreads();

    const int wave = tid >> 5, lane = tid & 31;
    const int n = lane & 15, h = (lane >> 4) & 1;
    const int dbase = wave * 128;             // this wave's 128 output columns

    const v8f vz = {0.f, 0.f, 0.f, 0.f, 0.f, 0.f, 0.f, 0.f};
    v8f acc[8];
    for (int j = 0; j < 8; ++j) acc[j] = vz;

    for (int s0 = 0; s0 <= t_hi; s0 += 256) {
        // ---- phase 1: two 16x16 score tiles per wave (keys sb0.. and sb0+16..) ----
        const int sb0 = s0 + wave * 32;
        v8f sc0 = vz, sc1 = vz;
        if (sb0 <= t_hi) {
            const unsigned short* wkrow0 = wkb + ((size_t)b * T_ + sb0) * D_;
            const unsigned short* wkrow1 = wkrow0 + (size_t)16 * D_;
            if (s0 + 256 <= t_hi)  // prefetch next key chunk
                __builtin_prefetch(wkb + ((size_t)b * T_ + sb0 + 256) * D_ + (size_t)lane * 64, 0, 0);
            for (int k = 0; k < D_; k += 32) {
                v16bf a   = ld_afrag(qs, D_, k, lane);
                v16bf bf0 = ld_bfrag(wkrow0, D_, k, lane);
                v16bf bf1 = ld_bfrag(wkrow1, D_, k, lane);
                sc0 = wmma_bf16(a, bf0, sc0);
                sc1 = wmma_bf16(a, bf1, sc1);
            }
        }
        // causal mask, bf16 stash to LDS, denominator row-sums
        for (int r = 0; r < 8; ++r) {
            const int row = r + h * 8;
            const int t_g = t0 + row;
            const float v0 = (sb0 + n      <= t_g) ? sc0[r] : 0.f;
            const float v1 = (sb0 + 16 + n <= t_g) ? sc1[r] : 0.f;
            ss[row * 256 + wave * 32 + n]      = f2bf(v0);
            ss[row * 256 + wave * 32 + 16 + n] = f2bf(v1);
            float rs = v0 + v1;
            rs += __shfl_xor(rs, 1, 32);
            rs += __shfl_xor(rs, 2, 32);
            rs += __shfl_xor(rs, 4, 32);
            rs += __shfl_xor(rs, 8, 32);
            if (n == 0) atomicAdd(&denom_s[row], rs);
        }
        __syncthreads();

        // ---- phase 2: scores @ v for this wave's 128 columns ----
        for (int sc8 = 0; sc8 < 8; ++sc8) {
            v16bf a = ld_afrag(ss, 256, sc8 * 32, lane);
            const int s_off = s0 + sc8 * 32;
            #pragma unroll
            for (int j = 0; j < 8; ++j) {
                const unsigned short* vcol = vt + ((size_t)b * D_ + dbase + j * 16) * T_;
                v16bf bf = ld_bfrag(vcol, T_, s_off, lane);
                acc[j] = wmma_bf16(a, bf, acc[j]);
            }
        }
        __syncthreads();
    }

    // ---- epilogue: divide by denominator ----
    for (int r = 0; r < 8; ++r) {
        const int row = r + h * 8;
        const int t_g = t0 + row;
        const float inv = 1.f / (denom_s[row] + EPS_);
        for (int j = 0; j < 8; ++j) {
            out[((size_t)b * T_ + t_g) * D_ + dbase + j * 16 + n] = acc[j][r] * inv;
        }
    }
}

// ---------------- Host launch ----------------
extern "C" void kernel_launch(void* const* d_in, const int* in_sizes, int n_in,
                              void* d_out, int out_size, void* d_ws, size_t ws_size,
                              hipStream_t stream) {
    (void)in_sizes; (void)n_in; (void)out_size; (void)ws_size;
    const float* x  = (const float*)d_in[0];
    const float* Wq = (const float*)d_in[1]; const float* bq = (const float*)d_in[2];
    const float* Wk = (const float*)d_in[3]; const float* bk = (const float*)d_in[4];
    const float* Wv = (const float*)d_in[5]; const float* bv = (const float*)d_in[6];
    const float* Wa = (const float*)d_in[7]; const float* ba = (const float*)d_in[8];
    float* out = (float*)d_out;

    const size_t NBT = (size_t)B_ * T_ * D_;      // 16,777,216 elements
    const size_t NW  = (size_t)D_ * D_;           // 1,048,576 elements

    char* p = (char*)d_ws;
    size_t off = 0;
    auto take = [&](size_t bytes) -> void* {
        void* r = p + off;
        off = (off + bytes + 255) & ~(size_t)255;
        return r;
    };
    unsigned short* xb  = (unsigned short*)take(NBT * 2);
    unsigned short* Wqb = (unsigned short*)take(NW * 2);
    unsigned short* Wkb = (unsigned short*)take(NW * 2);
    unsigned short* Wvb = (unsigned short*)take(NW * 2);
    unsigned short* Wab = (unsigned short*)take(NW * 2);
    unsigned short* q_b = (unsigned short*)take(NBT * 2);
    float*          kf  = (float*)take(NBT * 4);
    unsigned short* vtb = (unsigned short*)take(NBT * 2);
    float*          la  = (float*)take(NBT * 4);
    unsigned short* wkb = (unsigned short*)take(NBT * 2);

    // Stage 0: convert to bf16
    cvt_f32_bf16_kernel<<<4096, 256, 0, stream>>>(x,  xb,  (int)NBT);
    cvt_f32_bf16_kernel<<<1024, 256, 0, stream>>>(Wq, Wqb, (int)NW);
    cvt_f32_bf16_kernel<<<1024, 256, 0, stream>>>(Wk, Wkb, (int)NW);
    cvt_f32_bf16_kernel<<<1024, 256, 0, stream>>>(Wv, Wvb, (int)NW);
    cvt_f32_bf16_kernel<<<1024, 256, 0, stream>>>(Wa, Wab, (int)NW);

    // Stage 1: projections (grid.z = which projection)
    proj_kernel<<<dim3(B_ * T_ / 16, D_ / 512, 4), 256, 0, stream>>>(
        xb, Wqb, Wkb, Wvb, Wab, bq, bk, bv, ba, q_b, kf, vtb, la);

    // Stage 2: reverse gate scan
    scan_kernel<<<dim3(D_ / 256, B_), 256, 0, stream>>>(kf, la, wkb);

    // Stage 3: causal attention
    attn_kernel<<<dim3(T_ / 16, B_), 256, 0, stream>>>(q_b, wkb, vtb, out);
}
